// DecoderCell_64639257805371
// MI455X (gfx1250) — matrix operand order repaired and checked by reading, not compiled, not run on hardware
//
#include <hip/hip_runtime.h>
#include <hip/hip_bf16.h>

#define MASK_FILL_F (-1000.0f)

typedef __attribute__((ext_vector_type(16))) __bf16 bf16x16;
typedef __attribute__((ext_vector_type(8)))  float  f32x8;

// ---- f32 -> bf16 via native hardware converter (v_cvt_*_bf16_f32) ----
__device__ __forceinline__ bf16x16 pack16(const float* t) {
  bf16x16 r;
#pragma unroll
  for (int i = 0; i < 16; ++i) r[i] = (__bf16)t[i];
  return r;
}

// A-fragment: 16x32 bf16 tile of A (MxK, K-contiguous).
// Lane L<16 : row L, K = k0+0..7  and k0+16..23
// Lane L>=16: row L-16, K = k0+8..15 and k0+24..31
__device__ __forceinline__ bf16x16 load_a_frag(const float* __restrict__ A, int ldk,
                                               int rowbase, int k0, int lane) {
  const float* p = A + (size_t)(rowbase + (lane & 15)) * ldk + k0 + ((lane & 16) ? 8 : 0);
  float t[16];
  *(float4*)(t + 0)  = *(const float4*)(p + 0);
  *(float4*)(t + 4)  = *(const float4*)(p + 4);
  *(float4*)(t + 8)  = *(const float4*)(p + 16);
  *(float4*)(t + 12) = *(const float4*)(p + 20);
  return pack16(t);
}

// B-fragment: 32x16 bf16 tile where B-column n == row n0+n of W (NxK, K-contiguous).
// Lane L<16 : col L, K = k0+0..7 and k0+16..23 ; Lane L>=16: col L-16, K = k0+8..15 and k0+24..31
__device__ __forceinline__ bf16x16 load_b_frag(const float* __restrict__ W, int ldk,
                                               int n0, int k0, int lane) {
  const float* p = W + (size_t)(n0 + (lane & 15)) * ldk + k0 + ((lane & 16) ? 8 : 0);
  float t[16];
  *(float4*)(t + 0)  = *(const float4*)(p + 0);
  *(float4*)(t + 4)  = *(const float4*)(p + 4);
  *(float4*)(t + 8)  = *(const float4*)(p + 16);
  *(float4*)(t + 12) = *(const float4*)(p + 20);
  return pack16(t);
}

__device__ __forceinline__ f32x8 wmma_bf16(bf16x16 a, bf16x16 b, f32x8 c) {
  return __builtin_amdgcn_wmma_f32_16x16x32_bf16(false, a, false, b, (short)0, c, false, false);
}

// One K-segment of the 64xN GEMM: 4 M-tiles x 2 N-tiles per wave.
// Per K-step: 4 A-frag + 2 B-frag conversions feed 8 WMMAs.
__device__ __forceinline__ void gemm_seg42(const float* __restrict__ A,
                                           const float* __restrict__ W,
                                           int K, int n0, int lane, f32x8 acc[4][2]) {
  for (int k0 = 0; k0 < K; k0 += 32) {
    bf16x16 a[4];
#pragma unroll
    for (int mt = 0; mt < 4; ++mt) a[mt] = load_a_frag(A, K, mt * 16, k0, lane);
#pragma unroll
    for (int nt = 0; nt < 2; ++nt) {
      const float* wp = W + (size_t)(n0 + nt * 16 + (lane & 15)) * K + k0;
      if (k0 + 32 < K) __builtin_prefetch(wp + 32, 0, 1);   // global_prefetch_b8
      bf16x16 b = load_b_frag(W, K, n0 + nt * 16, k0, lane);
#pragma unroll
      for (int mt = 0; mt < 4; ++mt)
        acc[mt][nt] = wmma_bf16(a[mt], b, acc[mt][nt]);
    }
  }
}

// ---------------------------------------------------------------------------
// Generic fused GEMM: out[64 x N] = act( A1@W1^T + A2@W2^T + A3@W3^T + bias )
// block = 128 threads (4 waves); wave w owns cols [bx*128 + w*32, +32), rows 0..63.
// Every weight element is loaded & converted exactly once.
// act: 0=none, 1=sigmoid, 2=tanh
// ---------------------------------------------------------------------------
__global__ void __launch_bounds__(128)
dc_gemm3_kernel(const float* __restrict__ A1, const float* __restrict__ W1, int K1,
                const float* __restrict__ A2, const float* __restrict__ W2, int K2,
                const float* __restrict__ A3, const float* __restrict__ W3, int K3,
                const float* __restrict__ bias, float* __restrict__ out,
                int N, int act) {
  int lane = threadIdx.x & 31;
  int wave = threadIdx.x >> 5;
  int n0 = blockIdx.x * 128 + wave * 32;

  f32x8 acc[4][2] = {};
  gemm_seg42(A1, W1, K1, n0, lane, acc);
  if (A2) gemm_seg42(A2, W2, K2, n0, lane, acc);
  if (A3) gemm_seg42(A3, W3, K3, n0, lane, acc);

  int rsel = (lane & 16) ? 8 : 0;
#pragma unroll
  for (int nt = 0; nt < 2; ++nt) {
    int col = n0 + nt * 16 + (lane & 15);
    float bv = bias ? bias[col] : 0.0f;
#pragma unroll
    for (int mt = 0; mt < 4; ++mt) {
#pragma unroll
      for (int v = 0; v < 8; ++v) {
        int m = mt * 16 + v + rsel;
        float xv = acc[mt][nt][v] + bv;
        if (act == 1) xv = 1.0f / (1.0f + __expf(-xv));
        else if (act == 2) xv = tanhf(xv);
        out[(size_t)m * N + col] = xv;
      }
    }
  }
}

// ---------------------------------------------------------------------------
// Attention scores: e[l,b] = sum_a va[a]*tanh(dec[b,a] + (mask[l,b]?proj:-1000))
// proj = enc_flat(8192x2048) @ Ua(512x2048)^T.
// Block covers 64 rows x 512 cols: 8 waves, each 4 M-tiles x 4 N-tiles.
// ---------------------------------------------------------------------------
__global__ void __launch_bounds__(256)
dc_attn_kernel(const float* __restrict__ enc, const float* __restrict__ Ua,
               const float* __restrict__ dec, const unsigned char* __restrict__ mask,
               const float* __restrict__ va, float* __restrict__ e) {
  __shared__ float red[8][64];
  int lane = threadIdx.x & 31;
  int wave = threadIdx.x >> 5;
  int f0 = blockIdx.x * 64;             // flattened (l*64 + b) row base; l = blockIdx.x
  const int K = 2048;

  f32x8 acc[4][4] = {};
  for (int k0 = 0; k0 < K; k0 += 32) {
    bf16x16 a[4];
#pragma unroll
    for (int mt = 0; mt < 4; ++mt) a[mt] = load_a_frag(enc, K, f0 + mt * 16, k0, lane);
#pragma unroll
    for (int nt = 0; nt < 4; ++nt) {
      bf16x16 b = load_b_frag(Ua, K, wave * 64 + nt * 16, k0, lane);
#pragma unroll
      for (int mt = 0; mt < 4; ++mt)
        acc[mt][nt] = wmma_bf16(a[mt], b, acc[mt][nt]);
    }
  }

  int rsel = (lane & 16) ? 8 : 0;
#pragma unroll
  for (int mt = 0; mt < 4; ++mt) {
    float partial[8];
#pragma unroll
    for (int v = 0; v < 8; ++v) partial[v] = 0.0f;
#pragma unroll
    for (int nt = 0; nt < 4; ++nt) {
      int col = wave * 64 + nt * 16 + (lane & 15);
      float vav = va[col];
#pragma unroll
      for (int v = 0; v < 8; ++v) {
        int b_ = mt * 16 + v + rsel;           // row within block == batch index b
        float val = mask[f0 + b_] ? acc[mt][nt][v] : MASK_FILL_F;
        partial[v] += vav * tanhf(dec[b_ * 512 + col] + val);
      }
    }
    // butterfly reduce over the 16-lane half-wave (columns)
#pragma unroll
    for (int off = 1; off < 16; off <<= 1)
#pragma unroll
      for (int v = 0; v < 8; ++v) partial[v] += __shfl_xor(partial[v], off, 32);
    if ((lane & 15) == 0) {
#pragma unroll
      for (int v = 0; v < 8; ++v) red[wave][mt * 16 + v + rsel] = partial[v];
    }
  }
  __syncthreads();
  if (threadIdx.x < 64) {
    float s = 0.0f;
#pragma unroll
    for (int w = 0; w < 8; ++w) s += red[w][threadIdx.x];
    e[f0 + threadIdx.x] = s;
  }
}

// softmax over l (axis 0) of e(128x64); one thread per b
__global__ void dc_softmax_kernel(const float* __restrict__ e, float* __restrict__ alpha) {
  int b = threadIdx.x;
  float mx = -3.0e38f;
  for (int l = 0; l < 128; ++l) mx = fmaxf(mx, e[l * 64 + b]);
  float s = 0.0f;
  for (int l = 0; l < 128; ++l) s += __expf(e[l * 64 + b] - mx);
  float inv = 1.0f / s;
  for (int l = 0; l < 128; ++l) alpha[l * 64 + b] = __expf(e[l * 64 + b] - mx) * inv;
}

// c[b,h] = sum_l alpha[l,b] * enc[l,b,h] ; one block per b (2nd enc pass hits L2)
__global__ void __launch_bounds__(256)
dc_context_kernel(const float* __restrict__ alpha, const float* __restrict__ enc,
                  float* __restrict__ c) {
  int b = blockIdx.x;
  int h = threadIdx.x;
  float acc[8] = {};
  for (int l = 0; l < 128; ++l) {
    float a = alpha[l * 64 + b];
    const float* row = enc + (size_t)(l * 64 + b) * 2048;
#pragma unroll
    for (int j = 0; j < 8; ++j) acc[j] += a * row[h + j * 256];
  }
#pragma unroll
  for (int j = 0; j < 8; ++j) c[(size_t)b * 2048 + h + j * 256] = acc[j];
}

__global__ void dc_mul_kernel(const float* __restrict__ a, const float* __restrict__ b,
                              float* __restrict__ o, int n) {
  int i = blockIdx.x * blockDim.x + threadIdx.x;
  if (i < n) o[i] = a[i] * b[i];
}

// s = z*s_prop + (1-z)*sprev
__global__ void dc_scomb_kernel(const float* __restrict__ z, const float* __restrict__ sp,
                                const float* __restrict__ sprev, float* __restrict__ s, int n) {
  int i = blockIdx.x * blockDim.x + threadIdx.x;
  if (i < n) s[i] = z[i] * sp[i] + (1.0f - z[i]) * sprev[i];
}

// maxout k=2 over last dim: t[i] = max(tt[2i], tt[2i+1])
__global__ void dc_maxout_kernel(const float* __restrict__ tt, float* __restrict__ t, int n) {
  int i = blockIdx.x * blockDim.x + threadIdx.x;
  if (i < n) t[i] = fmaxf(tt[2 * i], tt[2 * i + 1]);
}

extern "C" void kernel_launch(void* const* d_in, const int* in_sizes, int n_in,
                              void* d_out, int out_size, void* d_ws, size_t ws_size,
                              hipStream_t stream) {
  (void)in_sizes; (void)n_in; (void)out_size; (void)ws_size;
  const float* x      = (const float*)d_in[0];
  const float* sprev  = (const float*)d_in[1];
  const float* enc    = (const float*)d_in[2];
  const unsigned char* mask = (const unsigned char*)d_in[3];
  const float* Ws_ = (const float*)d_in[4];
  const float* Wz_ = (const float*)d_in[5];
  const float* Wr_ = (const float*)d_in[6];
  const float* Us_ = (const float*)d_in[7];
  const float* Uz_ = (const float*)d_in[8];
  const float* Ur_ = (const float*)d_in[9];
  const float* Cs_ = (const float*)d_in[10];
  const float* Cz_ = (const float*)d_in[11];
  const float* Cr_ = (const float*)d_in[12];
  const float* bs_ = (const float*)d_in[13];
  const float* bz_ = (const float*)d_in[14];
  const float* br_ = (const float*)d_in[15];
  const float* va_ = (const float*)d_in[16];
  const float* Wa_ = (const float*)d_in[17];
  const float* Ua_ = (const float*)d_in[18];
  const float* Uo_ = (const float*)d_in[19];
  const float* Vo_ = (const float*)d_in[20];
  const float* Co_ = (const float*)d_in[21];
  const float* Wo_ = (const float*)d_in[22];

  float* ws    = (float*)d_ws;
  float* dec   = ws;              // 64*512   = 32768
  float* e     = ws + 32768;      // 128*64   = 8192
  float* alpha = ws + 40960;      // 128*64   = 8192
  float* ctx   = ws + 49152;      // 64*2048  = 131072
  float* r     = ws + 180224;     // 64*1024
  float* z     = ws + 245760;     // 64*1024
  float* rs    = ws + 311296;     // 64*1024
  float* sprop = ws + 376832;     // 64*1024
  float* tt    = ws + 442368;     // 64*1024
  float* t     = ws + 507904;     // 64*512

  float* s_out  = (float*)d_out;          // (64,1024)
  float* logits = s_out + 64 * 1024;      // (64,32000)

  // 1) dec_score = sprev @ Wa^T   (64x512, K=1024)
  dc_gemm3_kernel<<<dim3(4), 128, 0, stream>>>(
      sprev, Wa_, 1024, nullptr, nullptr, 0, nullptr, nullptr, 0,
      nullptr, dec, 512, 0);

  // 2) e[l,b] via fused enc_proj GEMM + mask + tanh + va-dot (one block per l)
  dc_attn_kernel<<<128, 256, 0, stream>>>(enc, Ua_, dec, mask, va_, e);

  // 3) alpha = softmax over l
  dc_softmax_kernel<<<1, 64, 0, stream>>>(e, alpha);

  // 4) context c = sum_l alpha*enc  (64x2048)
  dc_context_kernel<<<64, 256, 0, stream>>>(alpha, enc, ctx);

  // 5) r = sigmoid(xWr^T + sprev Ur^T + c Cr^T + br)
  dc_gemm3_kernel<<<dim3(8), 128, 0, stream>>>(
      x, Wr_, 512, sprev, Ur_, 1024, ctx, Cr_, 2048, br_, r, 1024, 1);
  // 6) z = sigmoid(...)
  dc_gemm3_kernel<<<dim3(8), 128, 0, stream>>>(
      x, Wz_, 512, sprev, Uz_, 1024, ctx, Cz_, 2048, bz_, z, 1024, 1);

  // 7) rs = r * sprev
  dc_mul_kernel<<<256, 256, 0, stream>>>(r, sprev, rs, 65536);

  // 8) s_prop = tanh(xWs^T + rs Us^T + c Cs^T + bs)
  dc_gemm3_kernel<<<dim3(8), 128, 0, stream>>>(
      x, Ws_, 512, rs, Us_, 1024, ctx, Cs_, 2048, bs_, sprop, 1024, 2);

  // 9) s = z*s_prop + (1-z)*sprev  -> d_out[0:65536]
  dc_scomb_kernel<<<256, 256, 0, stream>>>(z, sprop, sprev, s_out, 65536);

  // 10) t_tilde = sprev Uo^T + x Vo^T + c Co^T  (64x1024)
  dc_gemm3_kernel<<<dim3(8), 128, 0, stream>>>(
      sprev, Uo_, 1024, x, Vo_, 512, ctx, Co_, 2048, nullptr, tt, 1024, 0);

  // 11) maxout k=2 -> t (64x512)
  dc_maxout_kernel<<<128, 256, 0, stream>>>(tt, t, 32768);

  // 12) logits = t @ Wo^T  (64x32000, K=512)
  dc_gemm3_kernel<<<dim3(250), 128, 0, stream>>>(
      t, Wo_, 512, nullptr, nullptr, 0, nullptr, nullptr, 0,
      nullptr, logits, 32000, 0);
}
